// QRNNLayer_19662360281094
// MI455X (gfx1250) — matrix operand order repaired and checked by reading, not compiled, not run on hardware
//
#include <hip/hip_runtime.h>

// ---------------------------------------------------------------------------
// QRNN layer for MI455X (gfx1250, wave32, WMMA + async global->LDS copies).
//   y = inp @ W^T + b  -> split(z,f,o) -> tanh/sigmoid -> forget-mult scan
// GEMM: M=32768, N=3072, K=1024 (~206 GFLOP) via v_wmma_f32_16x16x32_bf16,
// K-loop double-buffered through LDS with global_load_async_to_lds_b128.
// ---------------------------------------------------------------------------

typedef __attribute__((ext_vector_type(16))) __bf16 v16bf;
typedef __attribute__((ext_vector_type(8)))  float  v8f;

constexpr int B_  = 16;
constexpr int S_  = 2048;
constexpr int D_  = 1024;
constexpr int H_  = 1024;
constexpr int M_  = B_ * S_;   // 32768 rows
constexpr int K_  = D_;        // 1024
constexpr int N3_ = 3 * H_;    // 3072 cols (z|f|o gates)

constexpr int TM  = 128;       // block tile M
constexpr int TN  = 128;       // block tile N
constexpr int TK  = 32;        // K step = WMMA K
constexpr int LDT = 48;        // LDS row stride in bf16 (96B: 16B-aligned)

union FragU { uint4 u[2]; v16bf v; };

__device__ __forceinline__ unsigned short bf16_rne(float f) {
  unsigned int u = __float_as_uint(f);
  unsigned int r = u + 0x7FFFu + ((u >> 16) & 1u);
  return (unsigned short)(r >> 16);
}
__device__ __forceinline__ float fast_sigmoid(float x) {
  return 1.0f / (1.0f + __expf(-x));
}
__device__ __forceinline__ float fast_tanh(float x) {
  return 1.0f - 2.0f / (__expf(2.0f * x) + 1.0f);
}

// CDNA5 async copy: memory -> LDS, no VGPR round trip, tracked by ASYNCcnt.
// VDST operand holds the workgroup-relative LDS byte offset (low 32 bits of a
// generic pointer to __shared__), VADDR is the 64-bit global address.
__device__ __forceinline__ void async_copy_b128(unsigned lds_off,
                                                const void* gptr) {
  asm volatile("global_load_async_to_lds_b128 %0, %1, off"
               :: "v"(lds_off), "v"((unsigned long long)(uintptr_t)gptr)
               : "memory");
}
__device__ __forceinline__ void wait_async0() {
  asm volatile("s_wait_asynccnt 0x0" ::: "memory");
}

// -------------------------- fp32 -> bf16 convert ---------------------------
__global__ void f32_to_bf16_kernel(const float* __restrict__ src,
                                   unsigned short* __restrict__ dst, int n4) {
  int i = blockIdx.x * blockDim.x + threadIdx.x;
  if (i >= n4) return;
  float4 v = ((const float4*)src)[i];
  ushort4 o;
  o.x = bf16_rne(v.x);
  o.y = bf16_rne(v.y);
  o.z = bf16_rne(v.z);
  o.w = bf16_rne(v.w);
  ((ushort4*)dst)[i] = o;
}

// ------------------- GEMM + bias + gate activation (WMMA) ------------------
// grid = (M/128, 3H/128); block = 256 threads = 8 waves in a 4(M) x 2(N) grid.
// Each wave: 32x64 output = 2x4 tiles of 16x16, K-loop of 32-wide bf16 WMMAs.
__global__ void __launch_bounds__(256)
qrnn_gemm_act_kernel(const unsigned short* __restrict__ Abf,
                     const unsigned short* __restrict__ Wbf,
                     const float* __restrict__ bias,
                     float* __restrict__ Zg,
                     float* __restrict__ Fg,
                     float* __restrict__ Og) {
  __shared__ __attribute__((aligned(16))) unsigned short sA[2][TM * LDT];
  __shared__ __attribute__((aligned(16))) unsigned short sB[2][TN * LDT];

  const int tid  = threadIdx.x;
  const int lane = tid & 31;
  const int wave = tid >> 5;
  const int wm   = wave & 3;   // wave position in M (0..3)
  const int wn   = wave >> 2;  // wave position in N (0..1)
  const int m0   = blockIdx.x * TM;
  const int n0   = blockIdx.y * TN;

  // Per-thread copy coordinates: each thread moves 2x16B chunks per tile.
  // Chunk j covers row j>>2, 16B-piece j&3; thread owns j=tid and j=tid+256.
  const int row0 = tid >> 2;
  const int c4   = (tid & 3) * 8;  // element offset within the 32-elem row
  constexpr unsigned BUFB = TM * LDT * sizeof(unsigned short);  // 12288 B
  const unsigned ldsA  = (unsigned)(uintptr_t)&sA[0][0];
  const unsigned ldsB  = (unsigned)(uintptr_t)&sB[0][0];
  const unsigned offL0 = (unsigned)((row0 * LDT + c4) * 2);
  const unsigned offL1 = (unsigned)(((row0 + 64) * LDT + c4) * 2);
  const unsigned short* gA0 = Abf + (size_t)(m0 + row0) * K_ + c4;
  const unsigned short* gA1 = Abf + (size_t)(m0 + row0 + 64) * K_ + c4;
  const unsigned short* gB0 = Wbf + (size_t)(n0 + row0) * K_ + c4;
  const unsigned short* gB1 = Wbf + (size_t)(n0 + row0 + 64) * K_ + c4;

  auto stage = [&](int k0, int buf) {
    const unsigned bb = (unsigned)buf * BUFB;
    async_copy_b128(ldsA + bb + offL0, gA0 + k0);
    async_copy_b128(ldsA + bb + offL1, gA1 + k0);
    async_copy_b128(ldsB + bb + offL0, gB0 + k0);
    async_copy_b128(ldsB + bb + offL1, gB1 + k0);
  };

  v8f acc[2][4] = {};

  stage(0, 0);          // prologue: fill buffer 0
  wait_async0();
  __syncthreads();

  for (int k0 = 0; k0 < K_; k0 += TK) {
    const int cur = (k0 >> 5) & 1;
    if (k0 + TK < K_) stage(k0 + TK, cur ^ 1);  // overlap copy with math

    const unsigned short* a = &sA[cur][0];
    const unsigned short* b = &sB[cur][0];

    // A fragment (16x32 bf16): lane<16 holds row M=lane, K 0-15;
    // lane>=16 holds same rows, K 16-31 (matches B's contiguous order).
    FragU fa[2];
    const int akoff = (lane >> 4) ? 16 : 0;
#pragma unroll
    for (int i = 0; i < 2; ++i) {
      const int arow = wm * 32 + i * 16 + (lane & 15);
      fa[i].u[0] = *(const uint4*)(&a[arow * LDT + akoff]);
      fa[i].u[1] = *(const uint4*)(&a[arow * LDT + akoff + 8]);
    }
    // B fragment (32x16 bf16): lane<16 holds col N=lane, K 0-15;
    // lane>=16 holds K 16-31 (one contiguous 32B run per lane).
    FragU fb[4];
    const int bkoff = (lane >> 4) ? 16 : 0;
#pragma unroll
    for (int j = 0; j < 4; ++j) {
      const int brow = wn * 64 + j * 16 + (lane & 15);
      fb[j].u[0] = *(const uint4*)(&b[brow * LDT + bkoff]);
      fb[j].u[1] = *(const uint4*)(&b[brow * LDT + bkoff + 8]);
    }

#pragma unroll
    for (int i = 0; i < 2; ++i)
#pragma unroll
      for (int j = 0; j < 4; ++j)
        acc[i][j] = __builtin_amdgcn_wmma_f32_16x16x32_bf16(
            false, fa[i].v, false, fb[j].v, (short)0, acc[i][j], false, false);

    wait_async0();      // this wave's copies into buf cur^1 are in LDS
    __syncthreads();    // all waves' copies visible; all reads of cur done
  }

  // Epilogue: bias + per-gate activation. The 128-wide N tile sits wholly in
  // one gate region (128 | 1024), so gate is block-uniform.
  const int gate = n0 >> 10;  // 0: z(tanh)  1: f(sigmoid)  2: o(sigmoid)
  float* __restrict__ Dst = (gate == 0) ? Zg : ((gate == 1) ? Fg : Og);
  const int mhalf = (lane >> 4) << 3;  // C/D layout: lanes>=16 hold M+8
#pragma unroll
  for (int i = 0; i < 2; ++i) {
    const int mbase = m0 + wm * 32 + i * 16 + mhalf;
#pragma unroll
    for (int j = 0; j < 4; ++j) {
      const int n    = n0 + wn * 64 + j * 16 + (lane & 15);
      const int hcol = n & (H_ - 1);
      const float bn = bias[n];
#pragma unroll
      for (int r = 0; r < 8; ++r) {
        const float v   = acc[i][j][r] + bn;
        const float res = (gate == 0) ? fast_tanh(v) : fast_sigmoid(v);
        Dst[(size_t)(mbase + r) * H_ + hcol] = res;
      }
    }
  }
}

// ------------------------- forget-mult scan over S -------------------------
// One thread per (b,h); consecutive h -> consecutive addresses (coalesced).
__global__ void qrnn_scan_kernel(const float* __restrict__ Zg,
                                 const float* __restrict__ Fg,
                                 const float* __restrict__ Og,
                                 float* __restrict__ out,
                                 float* __restrict__ c_last) {
  const int idx = blockIdx.x * blockDim.x + threadIdx.x;  // 0..B*H-1
  const int b   = idx >> 10;        // / H
  const int h   = idx & (H_ - 1);   // % H
  size_t p = (size_t)b * S_ * H_ + h;
  float c = 0.0f;
  for (int s = 0; s < S_; ++s, p += H_) {
    const float z = Zg[p];
    const float f = Fg[p];
    const float o = Og[p];
    c = fmaf(f, z - c, c);          // f*z + (1-f)*c
    out[p] = o * c;
    __builtin_prefetch(&Zg[p + 8 * (size_t)H_], 0, 1);  // speculative, no branch
  }
  c_last[idx] = c;
}

// ---------------------------------------------------------------------------
extern "C" void kernel_launch(void* const* d_in, const int* in_sizes, int n_in,
                              void* d_out, int out_size, void* d_ws, size_t ws_size,
                              hipStream_t stream) {
  const float* inp  = (const float*)d_in[0];  // [B,S,D]
  const float* W    = (const float*)d_in[1];  // [3H,D]
  const float* bias = (const float*)d_in[2];  // [3H]
  float* out = (float*)d_out;                 // [B,S,H] ++ [B,H]

  // Workspace layout: bf16 A (64MB) | bf16 W (6MB) | Z | F | O (128MB each)
  char* ws = (char*)d_ws;
  unsigned short* Abf = (unsigned short*)ws;
  size_t off = (size_t)M_ * K_ * sizeof(unsigned short);
  unsigned short* Wbf = (unsigned short*)(ws + off);
  off += (size_t)N3_ * K_ * sizeof(unsigned short);
  off = (off + 255) & ~(size_t)255;
  float* Zg = (float*)(ws + off); off += (size_t)M_ * H_ * sizeof(float);
  float* Fg = (float*)(ws + off); off += (size_t)M_ * H_ * sizeof(float);
  float* Og = (float*)(ws + off);

  {
    const int n4 = (M_ * K_) / 4;
    f32_to_bf16_kernel<<<(n4 + 255) / 256, 256, 0, stream>>>(inp, Abf, n4);
  }
  {
    const int n4 = (N3_ * K_) / 4;
    f32_to_bf16_kernel<<<(n4 + 255) / 256, 256, 0, stream>>>(W, Wbf, n4);
  }

  dim3 grid(M_ / TM, N3_ / TN);  // 256 x 24 workgroups
  qrnn_gemm_act_kernel<<<grid, 256, 0, stream>>>(Abf, Wbf, bias, Zg, Fg, Og);

  qrnn_scan_kernel<<<(B_ * H_) / 256, 256, 0, stream>>>(
      Zg, Fg, Og, out, out + (size_t)M_ * H_);
}
